// MS_Attention_linear_56702158242377
// MI455X (gfx1250) — compile-verified, matrix operand order checked by reading, not run on hardware
//
#include <hip/hip_runtime.h>

// ---------------- problem constants ----------------
#define TB 32          // T*B
#define CC 256
#define HW 3136        // 56*56
#define NHEAD 8
#define DH 32          // CC / NHEAD

typedef _Float16 h2  __attribute__((ext_vector_type(2)));
typedef _Float16 h8  __attribute__((ext_vector_type(8)));
typedef _Float16 h16 __attribute__((ext_vector_type(16)));
typedef float    f8  __attribute__((ext_vector_type(8)));
typedef int      i32x4 __attribute__((ext_vector_type(4)));

union Frag16 { h16 v; h8 h[2]; };

#if __has_builtin(__builtin_amdgcn_global_load_async_to_lds_b128) && \
    __has_builtin(__builtin_amdgcn_s_wait_asynccnt)
#define USE_ASYNC_LDS 1
#endif

// Load one 16x32 f16 WMMA operand fragment (A layout; B mirrored => identical
// per-lane addressing against a [row][K] array). p = row base + k-block start.
// lane<16: K = 0..7 (v0..3), 16..23 (v4..7); lane>=16: K = 8..15, 24..31.
__device__ __forceinline__ h16 load_frag(const _Float16* p, int hi) {
  Frag16 u;
  u.h[0] = *(const h8*)(p + hi * 8);
  u.h[1] = *(const h8*)(p + hi * 8 + 16);
  return u.v;
}

__device__ __forceinline__ float quant8(float x) {
  // clip(round(x),0,8)/8 ; rintf = round-to-nearest-even like jnp.round
  return fminf(fmaxf(rintf(x), 0.f), 8.f) * 0.125f;
}

#define WMMA(A, B, Cacc) \
  __builtin_amdgcn_wmma_f32_16x16x32_f16(false, (A), false, (B), (short)0, (Cacc), false, false)

// Stage `halfs` f16 (contiguous) from global into LDS, block-cooperatively.
// Uses gfx1250 async-to-LDS (ASYNCcnt) when the builtin exists.
__device__ __forceinline__ void stage_lds(const _Float16* __restrict__ src,
                                          _Float16* dst, int halfs,
                                          int tid, int nthreads) {
  int chunks = halfs >> 3;  // 16-byte chunks
#ifdef USE_ASYNC_LDS
  for (int c = tid; c < chunks; c += nthreads) {
    __builtin_amdgcn_global_load_async_to_lds_b128(
        (__attribute__((address_space(1))) i32x4*)(src + c * 8),
        (__attribute__((address_space(3))) i32x4*)(dst + c * 8), 0, 0);
  }
  __builtin_amdgcn_s_wait_asynccnt(0);
#else
  for (int c = tid; c < chunks; c += nthreads)
    *(h8*)(dst + c * 8) = *(const h8*)(src + c * 8);
#endif
}

// ---------------- 1) quantize input + transpose to pixel-major f16 ----------
// x: [TB, C, N] f32 (channel-major)  ->  xs: [TB, N, C] f16, values k/8
__global__ void k_quant_in(const float* __restrict__ x, _Float16* __restrict__ xs) {
  int g = blockIdx.x * blockDim.x + threadIdx.x;   // exactly TB*HW threads
  int tb = g / HW, n = g % HW;
  const float* src = x + (size_t)tb * CC * HW + n;
  _Float16* dst = xs + ((size_t)tb * HW + n) * CC;
  for (int c = 0; c < CC; c += 2) {
    float a = src[(size_t)c * HW];
    float b = src[(size_t)(c + 1) * HW];
    h2 pk;
    pk[0] = (_Float16)quant8(a);
    pk[1] = (_Float16)quant8(b);
    *(h2*)(dst + c) = pk;
  }
}

// ---------------- 2) weights -> f16, BN -> (scale, shift) ----------------
__global__ void k_prep(const float* __restrict__ qw, const float* __restrict__ kw,
                       const float* __restrict__ vw, const float* __restrict__ pw,
                       const float* __restrict__ qbn, const float* __restrict__ kbn,
                       const float* __restrict__ vbn, const float* __restrict__ pbn,
                       _Float16* __restrict__ wf, _Float16* __restrict__ pwh,
                       float* __restrict__ sf, float* __restrict__ tf,
                       float* __restrict__ sp, float* __restrict__ tp) {
  int g = blockIdx.x * blockDim.x + threadIdx.x;
  if (g < 768 * 256) {
    int o = g >> 8, c = g & 255;
    float w = (o < 256) ? qw[o * 256 + c]
            : (o < 512) ? kw[(o - 256) * 256 + c]
                        : vw[(o - 512) * 256 + c];
    wf[g] = (_Float16)w;
  } else if (g < 768 * 256 + 256 * 256) {
    int i = g - 768 * 256;
    pwh[i] = (_Float16)pw[i];
  } else {
    int i = g - (768 * 256 + 256 * 256);
    if (i < 768) {
      const float* bn = (i < 256) ? qbn : (i < 512) ? kbn : vbn;
      int c = i & 255;
      float s = bn[c] / sqrtf(bn[3 * 256 + c] + 1e-5f);
      sf[i] = s;
      tf[i] = bn[256 + c] - bn[2 * 256 + c] * s;
    } else if (i < 1024) {
      int c = i - 768;
      float s = pbn[c] / sqrtf(pbn[3 * 256 + c] + 1e-5f);
      sp[c] = s;
      tp[c] = pbn[256 + c] - pbn[2 * 256 + c] * s;
    }
  }
}

// ---------------- 3) fused QKV GEMM + BN + quant -------------------------
// Y[n, o] = sum_c xs[tb][n][c] * wf[o][c], o in [0,768)
// Block = 7 waves: one 16-pixel tile per wave, shared 64x256 weight tile in LDS.
// q -> qp [bh][n][e] pixel-major ; k -> kc [bh][d][n] ; v -> vc [bh][e][n]
__global__ void __launch_bounds__(224) k_qkv(const _Float16* __restrict__ xs,
                                             const _Float16* __restrict__ wf,
                                             const float* __restrict__ sf,
                                             const float* __restrict__ tf,
                                             _Float16* __restrict__ qp,
                                             _Float16* __restrict__ kc,
                                             _Float16* __restrict__ vc) {
  __shared__ _Float16 bt[64 * 256];  // 32 KB weight tile
  int tid = threadIdx.x;
  int wave = tid >> 5, lane = tid & 31, r = lane & 15, hi = lane >> 4;
  // grid = tb(32) x ng(28) x og(12)
  int og = blockIdx.x % 12;
  int ng = (blockIdx.x / 12) % 28;
  int tb = blockIdx.x / (12 * 28);
  int o0 = og * 64;
  int n0 = (ng * 7 + wave) * 16;

  stage_lds(wf + (size_t)o0 * CC, bt, 64 * CC, tid, 224);
  __syncthreads();

  const _Float16* arow = xs + ((size_t)tb * HW + n0 + r) * CC;
  f8 acc[4] = {};
#pragma unroll
  for (int k = 0; k < CC; k += 32) {
    h16 a = load_frag(arow + k, hi);
#pragma unroll
    for (int ot = 0; ot < 4; ++ot) {
      h16 b = load_frag(bt + (ot * 16 + r) * CC + k, hi);
      acc[ot] = WMMA(a, b, acc[ot]);
    }
  }
#pragma unroll
  for (int ot = 0; ot < 4; ++ot) {
    int o = o0 + ot * 16 + r;
    float s = sf[o], t = tf[o];
#pragma unroll
    for (int j = 0; j < 8; ++j) {
      int n = n0 + j + 8 * hi;
      _Float16 qv = (_Float16)quant8(acc[ot][j] * s + t);
      if (o < 256) {
        int h = o >> 5, e = o & 31;
        qp[(((size_t)tb * NHEAD + h) * HW + n) * DH + e] = qv;
      } else if (o < 512) {
        int oc = o - 256, h = oc >> 5, d = oc & 31;
        kc[(((size_t)tb * NHEAD + h) * DH + d) * HW + n] = qv;
      } else {
        int oc = o - 512, h = oc >> 5, e = oc & 31;
        vc[(((size_t)tb * NHEAD + h) * DH + e) * HW + n] = qv;
      }
    }
  }
}

// ---------------- 4) attention: kv = K^T V (reduce N), then O = Q kv -------
// one block (8 waves) per (tb, head)
__global__ void __launch_bounds__(256) k_attn(const _Float16* __restrict__ qp,
                                              const _Float16* __restrict__ kc,
                                              const _Float16* __restrict__ vc,
                                              _Float16* __restrict__ oq) {
  __shared__ float kv[32][32];
  __shared__ _Float16 kvT[32][32];
  int bh = blockIdx.x;  // tb*8 + h
  int tid = threadIdx.x;
  int wave = tid >> 5, lane = tid & 31, r = lane & 15, hi = lane >> 4;

  for (int i = tid; i < 1024; i += 256) ((float*)kv)[i] = 0.f;
  __syncthreads();

  const _Float16* kb = kc + (size_t)bh * DH * HW;
  const _Float16* vb = vc + (size_t)bh * DH * HW;

  f8 acc[2][2] = {};
  for (int kblk = wave; kblk < 98; kblk += 8) {  // 98 = HW/32
    int n = kblk * 32;
    h16 a0 = load_frag(kb + (size_t)r * HW + n, hi);
    h16 a1 = load_frag(kb + (size_t)(16 + r) * HW + n, hi);
    h16 b0 = load_frag(vb + (size_t)r * HW + n, hi);
    h16 b1 = load_frag(vb + (size_t)(16 + r) * HW + n, hi);
    acc[0][0] = WMMA(a0, b0, acc[0][0]);
    acc[0][1] = WMMA(a0, b1, acc[0][1]);
    acc[1][0] = WMMA(a1, b0, acc[1][0]);
    acc[1][1] = WMMA(a1, b1, acc[1][1]);
  }
#pragma unroll
  for (int dt = 0; dt < 2; ++dt)
#pragma unroll
    for (int et = 0; et < 2; ++et)
#pragma unroll
      for (int j = 0; j < 8; ++j)
        atomicAdd(&kv[dt * 16 + j + 8 * hi][et * 16 + r], acc[dt][et][j]);
  __syncthreads();

  // scale*2 = 2/sqrt(32); store transposed f16 so B-frag reads are contiguous
  for (int i = tid; i < 1024; i += 256) {
    int d = i >> 5, e = i & 31;
    kvT[e][d] = (_Float16)(kv[d][e] * 0.35355339059327373f);
  }
  __syncthreads();

  const _Float16* qb = qp + (size_t)bh * HW * DH;
  int h = bh & 7, tb = bh >> 3;
  for (int job = wave; job < 392; job += 8) {  // 196 n-tiles x 2 e-tiles
    int nt = job >> 1, et = job & 1;
    int n0 = nt * 16;
    h16 a = load_frag(qb + (size_t)(n0 + r) * DH, hi);
    h16 b = load_frag(&kvT[et * 16 + r][0], hi);
    f8 o = {};
    o = WMMA(a, b, o);
#pragma unroll
    for (int j = 0; j < 8; ++j) {
      int n = n0 + j + 8 * hi;
      int ch = h * DH + et * 16 + r;
      oq[((size_t)tb * HW + n) * CC + ch] = (_Float16)quant8(o[j]);
    }
  }
}

// ---------------- 5) projection GEMM + BN -> f32 output --------------------
// out[tb][o][n] = sum_c oq[tb][n][c] * pw[o][c] * sp[o] + tp[o]
__global__ void __launch_bounds__(224) k_proj(const _Float16* __restrict__ oq,
                                              const _Float16* __restrict__ pwh,
                                              const float* __restrict__ sp,
                                              const float* __restrict__ tp,
                                              float* __restrict__ out) {
  __shared__ _Float16 bt[64 * 256];  // 32 KB weight tile
  int tid = threadIdx.x;
  int wave = tid >> 5, lane = tid & 31, r = lane & 15, hi = lane >> 4;
  // grid = tb(32) x ng(28) x og(4)
  int og = blockIdx.x % 4;
  int ng = (blockIdx.x / 4) % 28;
  int tb = blockIdx.x / (4 * 28);
  int o0 = og * 64;
  int n0 = (ng * 7 + wave) * 16;

  stage_lds(pwh + (size_t)o0 * CC, bt, 64 * CC, tid, 224);
  __syncthreads();

  const _Float16* arow = oq + ((size_t)tb * HW + n0 + r) * CC;
  f8 acc[4] = {};
#pragma unroll
  for (int k = 0; k < CC; k += 32) {
    h16 a = load_frag(arow + k, hi);
#pragma unroll
    for (int ot = 0; ot < 4; ++ot) {
      h16 b = load_frag(bt + (ot * 16 + r) * CC + k, hi);
      acc[ot] = WMMA(a, b, acc[ot]);
    }
  }
#pragma unroll
  for (int ot = 0; ot < 4; ++ot) {
    int o = o0 + ot * 16 + r;
    float s = sp[o], t = tp[o];
#pragma unroll
    for (int j = 0; j < 8; ++j) {
      int n = n0 + j + 8 * hi;
      out[((size_t)tb * CC + o) * HW + n] = acc[ot][j] * s + t;
    }
  }
}

// ---------------- launcher ----------------
extern "C" void kernel_launch(void* const* d_in, const int* in_sizes, int n_in,
                              void* d_out, int out_size, void* d_ws, size_t ws_size,
                              hipStream_t stream) {
  const float* x   = (const float*)d_in[0];
  const float* qw  = (const float*)d_in[1];
  const float* qbn = (const float*)d_in[2];
  const float* kw  = (const float*)d_in[3];
  const float* kbn = (const float*)d_in[4];
  const float* vw  = (const float*)d_in[5];
  const float* vbn = (const float*)d_in[6];
  const float* pw  = (const float*)d_in[7];
  const float* pbn = (const float*)d_in[8];
  float* out = (float*)d_out;

  char* ws = (char*)d_ws;
  const size_t SZ_ACT = (size_t)TB * HW * CC * sizeof(_Float16);  // 51,380,224 B
  _Float16* xs  = (_Float16*)(ws + 0 * SZ_ACT);   // also reused for oq
  _Float16* qp  = (_Float16*)(ws + 1 * SZ_ACT);
  _Float16* kc  = (_Float16*)(ws + 2 * SZ_ACT);
  _Float16* vc  = (_Float16*)(ws + 3 * SZ_ACT);
  char* p = ws + 4 * SZ_ACT;
  _Float16* wf  = (_Float16*)p;           p += (size_t)768 * 256 * 2;
  _Float16* pwh = (_Float16*)p;           p += (size_t)256 * 256 * 2;
  float* sf = (float*)p;                  p += 768 * 4;
  float* tf = (float*)p;                  p += 768 * 4;
  float* sp = (float*)p;                  p += 256 * 4;
  float* tp = (float*)p;

  // 1) quantize + transpose input: TB*HW threads
  k_quant_in<<<(TB * HW) / 256, 256, 0, stream>>>(x, xs);
  // 2) weight/BN prep: 768*256 + 256*256 + 1024 = 263168 threads
  k_prep<<<263168 / 256, 256, 0, stream>>>(qw, kw, vw, pw, qbn, kbn, vbn, pbn,
                                           wf, pwh, sf, tf, sp, tp);
  // 3) QKV: 32 tb x 28 n-groups x 12 o-groups blocks, 7 waves each
  k_qkv<<<32 * 28 * 12, 224, 0, stream>>>(xs, wf, sf, tf, qp, kc, vc);
  // 4) attention: one block per (tb, head); writes oq into xs buffer
  k_attn<<<TB * NHEAD, 256, 0, stream>>>(qp, kc, vc, xs);
  // 5) projection: 32 tb x 28 n-groups x 4 o-groups blocks, 7 waves each
  k_proj<<<32 * 28 * 4, 224, 0, stream>>>(xs, pwh, sp, tp, out);
}
